// SwinTransformerBlock_MSWA_21002390077832
// MI455X (gfx1250) — matrix operand order repaired and checked by reading, not compiled, not run on hardware
//
#include <hip/hip_runtime.h>

// ---------------------------------------------------------------------------
// Problem constants (Swin MSWA block): B=8, H=W=56, C=384, L=3136
// ---------------------------------------------------------------------------
constexpr int kB  = 8;
constexpr int kH  = 56;
constexpr int kW  = 56;
constexpr int kC  = 384;
constexpr int kL  = 3136;          // H*W
constexpr int kM  = kB * kL;       // 25088 rows
constexpr int kNq = 3 * kC;        // 1152

typedef __attribute__((ext_vector_type(16))) __bf16 v16bf;
typedef __attribute__((ext_vector_type(8)))  float  v8f;

// ---------------------------------------------------------------------------
// f32 (KxN row-major) -> bf16 transposed (NxK row-major), done once per weight
// so the GEMM B-tile staging is fully vectorized (no per-tile LDS transpose).
// ---------------------------------------------------------------------------
__global__ void cvt_bf16_t_kernel(const float* __restrict__ in,
                                  __bf16* __restrict__ outT, int K, int N) {
  int idx = blockIdx.x * 256 + threadIdx.x;
  if (idx >= K * N) return;
  const int k = idx / N, n = idx - k * N;
  outT[(size_t)n * K + k] = (__bf16)in[idx];
}

// ---------------------------------------------------------------------------
// LayerNorm over C=384, one block (128 threads) per row, bf16 output
// ---------------------------------------------------------------------------
__global__ __launch_bounds__(128) void ln_bf16_kernel(
    const float* __restrict__ x, const float* __restrict__ g,
    const float* __restrict__ b, __bf16* __restrict__ out) {
  const int row = blockIdx.x, tid = threadIdx.x;
  const float* xr = x + (size_t)row * kC;
  float v0 = xr[tid], v1 = xr[tid + 128], v2 = xr[tid + 256];
  __shared__ float rs[128], rq[128];
  rs[tid] = v0 + v1 + v2;
  rq[tid] = v0 * v0 + v1 * v1 + v2 * v2;
  __syncthreads();
  for (int off = 64; off > 0; off >>= 1) {
    if (tid < off) { rs[tid] += rs[tid + off]; rq[tid] += rq[tid + off]; }
    __syncthreads();
  }
  const float mean = rs[0] * (1.0f / kC);
  const float var  = rq[0] * (1.0f / kC) - mean * mean;
  const float r    = rsqrtf(var + 1e-5f);
  __bf16* orow = out + (size_t)row * kC;
  orow[tid]       = (__bf16)((v0 - mean) * r * g[tid]       + b[tid]);
  orow[tid + 128] = (__bf16)((v1 - mean) * r * g[tid + 128] + b[tid + 128]);
  orow[tid + 256] = (__bf16)((v2 - mean) * r * g[tid + 256] + b[tid + 256]);
}

// ---------------------------------------------------------------------------
// WMMA bf16 GEMM: C(MxN) = A(MxK) * Bt^T + bias [+ epilogue]
//   A is MxK row-major, Bt is NxK row-major (pre-transposed weights).
//   Block tile 64x128, K-step 32, 8 waves each owning a 32x32 output tile
//   -> 4x v_wmma_f32_16x16x32_bf16 per wave per K step.
// EPI: 0 = none, 1 = add residual (f32), 2 = exact GELU
// OBF: 0 = f32 output, 1 = bf16 output
// ---------------------------------------------------------------------------
template <int EPI, int OBF>
__global__ __launch_bounds__(256) void gemm_wmma_kernel(
    const __bf16* __restrict__ A, const __bf16* __restrict__ Bt,
    const float* __restrict__ bias, const float* __restrict__ resid,
    float* __restrict__ Cf, __bf16* __restrict__ Cb,
    int M, int N, int K) {
  __shared__ __attribute__((aligned(16))) __bf16 sA[64][40];   // [m][k], padded
  __shared__ __attribute__((aligned(16))) __bf16 sB[128][40];  // [n][k], padded
  const int tid  = threadIdx.x;
  const int bn   = blockIdx.x * 128;
  const int bm   = blockIdx.y * 64;
  const int lane = tid & 31, wv = tid >> 5;
  const int wm   = (wv >> 2) * 32;   // wave M origin (0/32)
  const int wn   = (wv & 3) * 32;    // wave N origin (0..96)
  const int l16  = lane & 15, half = lane >> 4;

  v8f acc[2][2];
#pragma unroll
  for (int i = 0; i < 2; ++i)
#pragma unroll
    for (int j = 0; j < 2; ++j)
#pragma unroll
      for (int e = 0; e < 8; ++e) acc[i][j][e] = 0.0f;

  const int arow = tid >> 2, ak0 = (tid & 3) * 8;   // A: 64 rows x 32 k
  const int brow = tid >> 1, bk0 = (tid & 1) * 16;  // B: 128 n-rows x 32 k

  for (int kk = 0; kk < K; kk += 32) {
    __syncthreads();
    {
      // A tile: one 16B vector load/store per thread
      const uint4 va = *(const uint4*)(A + (size_t)(bm + arow) * K + kk + ak0);
      *(uint4*)&sA[arow][ak0] = va;
      // B tile (pre-transposed): two 16B vector loads/stores per thread
      const __bf16* gB = Bt + (size_t)(bn + brow) * K + kk + bk0;
      const uint4 p0 = *(const uint4*)gB;
      const uint4 p1 = *(const uint4*)(gB + 8);
      *(uint4*)&sB[brow][bk0]     = p0;
      *(uint4*)&sB[brow][bk0 + 8] = p1;
      if (kk + 32 < K) {
        __builtin_prefetch(A + (size_t)(bm + arow) * K + kk + 32 + ak0, 0, 1);
        __builtin_prefetch(Bt + (size_t)(bn + brow) * K + kk + 32 + bk0, 0, 1);
      }
    }
    __syncthreads();

    union Frag { uint4 q[2]; v16bf v; };
    Frag af[2], bfr[2];
#pragma unroll
    for (int mt = 0; mt < 2; ++mt) {
      // A 16-bit fragment: lane half selects +8 inside each 16-k block
      const int row = wm + mt * 16 + l16;
      af[mt].q[0] = *(const uint4*)&sA[row][half * 8];
      af[mt].q[1] = *(const uint4*)&sA[row][16 + half * 8];
    }
#pragma unroll
    for (int nt = 0; nt < 2; ++nt) {
      // B 16-bit fragment: lane half selects +8 inside each 16-k block
      const int col = wn + nt * 16 + l16;
      bfr[nt].q[0] = *(const uint4*)&sB[col][half * 8];
      bfr[nt].q[1] = *(const uint4*)&sB[col][16 + half * 8];
    }
#pragma unroll
    for (int mt = 0; mt < 2; ++mt)
#pragma unroll
      for (int nt = 0; nt < 2; ++nt)
        acc[mt][nt] = __builtin_amdgcn_wmma_f32_16x16x32_bf16(
            false, af[mt].v, false, bfr[nt].v, (short)0, acc[mt][nt],
            false, false);
  }

  // Epilogue. C/D layout: VGPR r -> row (r + 8*half), col = lane%16.
#pragma unroll
  for (int mt = 0; mt < 2; ++mt)
#pragma unroll
    for (int nt = 0; nt < 2; ++nt) {
      const int gn = bn + wn + nt * 16 + l16;
      const float bv = bias[gn];
#pragma unroll
      for (int r = 0; r < 8; ++r) {
        const int gm = bm + wm + mt * 16 + half * 8 + r;
        float v = acc[mt][nt][r] + bv;
        if (EPI == 1) v += resid[(size_t)gm * N + gn];
        if (EPI == 2) v = 0.5f * v * (1.0f + erff(v * 0.70710678118654752f));
        if (OBF) Cb[(size_t)gm * N + gn] = (__bf16)v;
        else     Cf[(size_t)gm * N + gn] = v;
      }
    }
}

// ---------------------------------------------------------------------------
// Gather one q/k/v element through the reference's reshape scramble.
//   chunk = ((b*nW + wh)*nW + ww)*3 + s ; f = t*128 + h*32 + d
//   g = s*ws*ws*128 + f indexes the contiguous (ws, ws, C) window block.
// ---------------------------------------------------------------------------
__device__ __forceinline__ float qkv_elem(const float* __restrict__ qkv,
                                          int b, int wh, int ww, int s,
                                          int ws, int f, int off) {
  const int g   = s * ws * ws * 128 + f;
  const int wsC = ws * kC;
  const int i   = g / wsC;
  int rem       = g - i * wsC;
  const int j   = rem / kC;
  const int ch  = rem - j * kC;
  const int hh  = wh * ws + i;
  const int w2  = ww * ws + j;
  if (hh >= kH || w2 >= kW) return 0.0f;   // zero padding
  return qkv[((size_t)(b * kL + hh * kW + w2)) * kNq + off + ch];
}

// ---------------------------------------------------------------------------
// Multi-scale window attention. grid = (chunks, 4 heads), 256 threads.
// K/V (n x 32) staged in LDS; per-wave row softmax with shfl reductions.
// Output written in `out`'s native flat layout: chunk*n*128 + t*128 + h*32 + d
// ---------------------------------------------------------------------------
__global__ __launch_bounds__(256) void mswa_kernel(
    const float* __restrict__ qkv, const float* __restrict__ table,
    __bf16* __restrict__ raw, int ws, int nW) {
  __shared__ float kbuf[121 * 32];
  __shared__ float vbuf[121 * 32];
  __shared__ float qrow[8][32];
  __shared__ float srow[8][128];
  const int n    = ws * ws;
  const int head = blockIdx.y;
  int c = blockIdx.x;
  const int s  = c % 3; c /= 3;
  const int ww = c % nW; c /= nW;
  const int wh = c % nW;
  const int b  = c / nW;
  const int tid = threadIdx.x, lane = tid & 31, wv = tid >> 5;
  const float scale = 0.17677669529663687f;  // 32^-0.5
  const int tws = 2 * ws - 1;

  for (int idx = tid; idx < n * 32; idx += 256) {
    const int m = idx >> 5, d = idx & 31;
    const int f = m * 128 + head * 32 + d;
    kbuf[idx] = qkv_elem(qkv, b, wh, ww, s, ws, f, kC);
    vbuf[idx] = qkv_elem(qkv, b, wh, ww, s, ws, f, 2 * kC);
  }
  __syncthreads();

  for (int t0 = 0; t0 < n; t0 += 8) {
    const int t = t0 + wv;
    float inv = 0.0f;
    if (t < n)
      qrow[wv][lane] = qkv_elem(qkv, b, wh, ww, s, ws,
                                t * 128 + head * 32 + lane, 0);
    __syncthreads();
    if (t < n) {
      const int ti = t / ws, tj = t - ti * ws;
      float sloc[4];
      int cnt = 0;
      for (int m = lane; m < n; m += 32, ++cnt) {
        float dot = 0.0f;
#pragma unroll
        for (int d = 0; d < 32; ++d) dot += qrow[wv][d] * kbuf[m * 32 + d];
        const int mi = m / ws, mj = m - mi * ws;
        const float bia =
            table[((ti - mi + ws - 1) * tws + (tj - mj + ws - 1)) * 4 + head];
        sloc[cnt] = dot * scale + bia;
      }
      float lmax = -3.4e38f;
      for (int ii = 0; ii < cnt; ++ii) lmax = fmaxf(lmax, sloc[ii]);
#pragma unroll
      for (int off = 16; off > 0; off >>= 1)
        lmax = fmaxf(lmax, __shfl_xor(lmax, off, 32));
      float lsum = 0.0f;
      {
        int m = lane;
        for (int ii = 0; ii < cnt; ++ii, m += 32) {
          const float e = expf(sloc[ii] - lmax);
          srow[wv][m] = e;
          lsum += e;
        }
      }
#pragma unroll
      for (int off = 16; off > 0; off >>= 1)
        lsum += __shfl_xor(lsum, off, 32);
      inv = 1.0f / lsum;
    }
    __syncthreads();
    if (t < n) {
      float o = 0.0f;
      for (int m = 0; m < n; ++m) o += srow[wv][m] * vbuf[m * 32 + lane];
      raw[(size_t)blockIdx.x * n * 128 + t * 128 + head * 32 + lane] =
          (__bf16)(o * inv);
    }
    __syncthreads();
  }
}

// ---------------------------------------------------------------------------
// Map attention outputs through the reference's reshape/transpose/crop into
// the fusion-GEMM input layout (B, L, 3*C), bf16.
// ---------------------------------------------------------------------------
__global__ void gather_fuse_kernel(const __bf16* __restrict__ r3,
                                   const __bf16* __restrict__ r7,
                                   const __bf16* __restrict__ r11,
                                   __bf16* __restrict__ fuse) {
  const long long total = (long long)kM * kNq;
  const long long idx = (long long)blockIdx.x * 256 + threadIdx.x;
  if (idx >= total) return;
  const int ccat = (int)(idx % kNq);
  const int rowi = (int)(idx / kNq);
  const int b = rowi / kL, l = rowi % kL;
  const int si = ccat / kC, cc = ccat % kC;
  int ws, nW;
  const __bf16* raw;
  if (si == 0)      { ws = 3;  nW = 19; raw = r3;  }
  else if (si == 1) { ws = 7;  nW = 8;  raw = r7;  }
  else              { ws = 11; nW = 6;  raw = r11; }
  const int p = l * kC + cc;                 // flat within (s, h, w, 128)
  const int s = p / (kL * 128);
  int rem = p - s * (kL * 128);
  const int h = rem / (kW * 128); rem -= h * (kW * 128);
  const int w = rem / 128;
  const int c128 = rem - w * 128;
  const int wh = h / ws, i = h - wh * ws;
  const int wwi = w / ws, j = w - wwi * ws;
  const long long F =
      (((((long long)(b * 3 + s) * nW + wh) * nW + wwi) * ws + i) * ws + j) *
          128 + c128;
  fuse[idx] = raw[F];
}

// ---------------------------------------------------------------------------
// Depthwise 3x3 conv (SAME, groups=C) on (B, L, C) f32 layout.
// ---------------------------------------------------------------------------
__global__ void dwconv_kernel(const float* __restrict__ m,
                              const float* __restrict__ w,
                              const float* __restrict__ bias,
                              float* __restrict__ y) {
  const int idx = blockIdx.x * 256 + threadIdx.x;
  if (idx >= kM * kC) return;
  const int ch = idx % kC;
  int pos = idx / kC;
  const int x = pos % kW; pos /= kW;
  const int h = pos % kH;
  const int b = pos / kH;
  float acc = bias[ch];
#pragma unroll
  for (int dh = 0; dh < 3; ++dh) {
    const int hh = h + dh - 1;
    if (hh < 0 || hh >= kH) continue;
#pragma unroll
    for (int dw = 0; dw < 3; ++dw) {
      const int w2 = x + dw - 1;
      if (w2 < 0 || w2 >= kW) continue;
      acc += m[((size_t)(b * kL) + hh * kW + w2) * kC + ch] *
             w[ch * 9 + dh * 3 + dw];
    }
  }
  y[idx] = acc;
}

// ---------------------------------------------------------------------------
// Final: out = h0 + m + LN(y; convn_g, convn_b)
// ---------------------------------------------------------------------------
__global__ __launch_bounds__(128) void final_kernel(
    const float* __restrict__ h0, const float* __restrict__ m,
    const float* __restrict__ y, const float* __restrict__ g,
    const float* __restrict__ b, float* __restrict__ out) {
  const int row = blockIdx.x, tid = threadIdx.x;
  const float* yr = y + (size_t)row * kC;
  float v0 = yr[tid], v1 = yr[tid + 128], v2 = yr[tid + 256];
  __shared__ float rs[128], rq[128];
  rs[tid] = v0 + v1 + v2;
  rq[tid] = v0 * v0 + v1 * v1 + v2 * v2;
  __syncthreads();
  for (int off = 64; off > 0; off >>= 1) {
    if (tid < off) { rs[tid] += rs[tid + off]; rq[tid] += rq[tid + off]; }
    __syncthreads();
  }
  const float mean = rs[0] * (1.0f / kC);
  const float var  = rq[0] * (1.0f / kC) - mean * mean;
  const float r    = rsqrtf(var + 1e-5f);
  const float* h0r = h0 + (size_t)row * kC;
  const float* mr  = m + (size_t)row * kC;
  float* outr      = out + (size_t)row * kC;
  outr[tid]       = h0r[tid]       + mr[tid]       + (v0 - mean) * r * g[tid]       + b[tid];
  outr[tid + 128] = h0r[tid + 128] + mr[tid + 128] + (v1 - mean) * r * g[tid + 128] + b[tid + 128];
  outr[tid + 256] = h0r[tid + 256] + mr[tid + 256] + (v2 - mean) * r * g[tid + 256] + b[tid + 256];
}

// ---------------------------------------------------------------------------
// Host launch
// ---------------------------------------------------------------------------
extern "C" void kernel_launch(void* const* d_in, const int* in_sizes, int n_in,
                              void* d_out, int out_size, void* d_ws,
                              size_t ws_size, hipStream_t stream) {
  (void)in_sizes; (void)n_in; (void)out_size; (void)ws_size;
  const float* x      = (const float*)d_in[0];
  const float* n1g    = (const float*)d_in[1];
  const float* n1b    = (const float*)d_in[2];
  const float* qkv_w  = (const float*)d_in[3];
  const float* qkv_b  = (const float*)d_in[4];
  const float* t3     = (const float*)d_in[5];
  const float* t7     = (const float*)d_in[6];
  const float* t11    = (const float*)d_in[7];
  const float* fus_w  = (const float*)d_in[8];
  const float* fus_b  = (const float*)d_in[9];
  const float* n2g    = (const float*)d_in[10];
  const float* n2b    = (const float*)d_in[11];
  const float* fc1_w  = (const float*)d_in[12];
  const float* fc1_b  = (const float*)d_in[13];
  const float* fc2_w  = (const float*)d_in[14];
  const float* fc2_b  = (const float*)d_in[15];
  const float* conv_w = (const float*)d_in[16];
  const float* conv_b = (const float*)d_in[17];
  const float* cng    = (const float*)d_in[18];
  const float* cnb    = (const float*)d_in[19];
  float* out = (float*)d_out;

  char* base = (char*)d_ws;
  size_t off = 0;
  auto alloc = [&](size_t bytes) -> void* {
    void* p = base + off;
    off += (bytes + 255) & ~(size_t)255;
    return p;
  };

  __bf16* w_qkv  = (__bf16*)alloc((size_t)kC * kNq * 2);       // N x K
  __bf16* w_fus  = (__bf16*)alloc((size_t)kNq * kC * 2);       // N x K
  __bf16* w_fc1  = (__bf16*)alloc((size_t)kC * 4 * kC * 2);    // N x K
  __bf16* w_fc2  = (__bf16*)alloc((size_t)4 * kC * kC * 2);    // N x K
  __bf16* xn     = (__bf16*)alloc((size_t)kM * kC * 2);
  float*  qkvbuf = (float*) alloc((size_t)kM * kNq * 4);
  __bf16* raw3   = (__bf16*)alloc((size_t)8664 * 9 * 128 * 2);
  __bf16* raw7   = (__bf16*)alloc((size_t)1536 * 49 * 128 * 2);
  __bf16* raw11  = (__bf16*)alloc((size_t)864 * 121 * 128 * 2);
  __bf16* fusein = (__bf16*)alloc((size_t)kM * kNq * 2);
  float*  h0     = (float*) alloc((size_t)kM * kC * 4);
  __bf16* xn2    = (__bf16*)alloc((size_t)kM * kC * 2);
  __bf16* hidden = (__bf16*)alloc((size_t)kM * 4 * kC * 2);
  float*  mbuf   = (float*) alloc((size_t)kM * kC * 4);
  float*  ybuf   = (float*) alloc((size_t)kM * kC * 4);

  // 1) weights -> bf16, pre-transposed to N x K (one-time cost)
  cvt_bf16_t_kernel<<<(kC * kNq + 255) / 256, 256, 0, stream>>>(qkv_w, w_qkv, kC, kNq);
  cvt_bf16_t_kernel<<<(kNq * kC + 255) / 256, 256, 0, stream>>>(fus_w, w_fus, kNq, kC);
  cvt_bf16_t_kernel<<<(kC * 4 * kC + 255) / 256, 256, 0, stream>>>(fc1_w, w_fc1, kC, 4 * kC);
  cvt_bf16_t_kernel<<<(4 * kC * kC + 255) / 256, 256, 0, stream>>>(fc2_w, w_fc2, 4 * kC, kC);

  // 2) LN1
  ln_bf16_kernel<<<kM, 128, 0, stream>>>(x, n1g, n1b, xn);

  // 3) QKV GEMM: (25088 x 384) * (384 x 1152)
  gemm_wmma_kernel<0, 0><<<dim3(kNq / 128, kM / 64), 256, 0, stream>>>(
      xn, w_qkv, qkv_b, nullptr, qkvbuf, nullptr, kM, kNq, kC);

  // 4) Multi-scale window attention
  mswa_kernel<<<dim3(8664, 4), 256, 0, stream>>>(qkvbuf, t3,  raw3,  3, 19);
  mswa_kernel<<<dim3(1536, 4), 256, 0, stream>>>(qkvbuf, t7,  raw7,  7, 8);
  mswa_kernel<<<dim3(864, 4),  256, 0, stream>>>(qkvbuf, t11, raw11, 11, 6);

  // 5) Scramble attention outputs into fusion input (B, L, 1152)
  gather_fuse_kernel<<<(unsigned)(((long long)kM * kNq + 255) / 256), 256, 0,
                       stream>>>(raw3, raw7, raw11, fusein);

  // 6) Fusion GEMM + residual: h0 = x + fusein * W + b
  gemm_wmma_kernel<1, 0><<<dim3(kC / 128, kM / 64), 256, 0, stream>>>(
      fusein, w_fus, fus_b, x, h0, nullptr, kM, kC, kNq);

  // 7) LN2
  ln_bf16_kernel<<<kM, 128, 0, stream>>>(h0, n2g, n2b, xn2);

  // 8) MLP fc1 + exact GELU -> bf16 hidden
  gemm_wmma_kernel<2, 1><<<dim3(4 * kC / 128, kM / 64), 256, 0, stream>>>(
      xn2, w_fc1, fc1_b, nullptr, nullptr, hidden, kM, 4 * kC, kC);

  // 9) MLP fc2 -> m (f32)
  gemm_wmma_kernel<0, 0><<<dim3(kC / 128, kM / 64), 256, 0, stream>>>(
      hidden, w_fc2, fc2_b, nullptr, mbuf, nullptr, kM, kC, 4 * kC);

  // 10) depthwise conv
  dwconv_kernel<<<(kM * kC + 255) / 256, 256, 0, stream>>>(mbuf, conv_w,
                                                           conv_b, ybuf);

  // 11) out = h0 + m + LN(y)
  final_kernel<<<kM, 128, 0, stream>>>(h0, mbuf, ybuf, cng, cnb, out);
}
